// GC_Att_14491219657381
// MI455X (gfx1250) — matrix operand order repaired and checked by reading, not compiled, not run on hardware
//
#include <hip/hip_runtime.h>
#include <hip/hip_bf16.h>
#include <cmath>
#include <cstdint>

// ---------------------------------------------------------------------------
// MI455X (gfx1250) implementation.
// Shapes: N=6000, V=3, F=1000(pad 1024), H=256, L=4, C=7.
// adj_i is maintained implicitly as g[n,v] coefficients over adj_views,
// so per-layer HBM traffic is one 432MB stream of adj_views; all GEMMs are
// v_wmma_f32_16x16x32_bf16 with f32 accumulation. The adj stream is moved
// HBM->LDS by the Tensor Data Mover (double-buffered, TENSORcnt-synced),
// with a VALU fallback when the TDM builtin is unavailable.
// ---------------------------------------------------------------------------

#define NN6000 6000
#define HH 256
#define VV 3
#define LL 4
#define CC 7
#define FP 1024
#define NPAD 6016   // 6016 = 376*16, B-operand column padding

typedef __attribute__((ext_vector_type(8)))  float  v8f;
typedef __attribute__((ext_vector_type(8)))  __bf16 v8bf;
typedef __attribute__((ext_vector_type(16))) __bf16 v16bf;
typedef __attribute__((ext_vector_type(4)))  unsigned int v4u;
typedef __attribute__((ext_vector_type(4)))  int v4i;
typedef __attribute__((ext_vector_type(8)))  int v8i;

union BF16x16 { v16bf v; v8bf h[2]; };

static __device__ __forceinline__ v16bf pack16(v8bf lo, v8bf hi) {
  BF16x16 u; u.h[0] = lo; u.h[1] = hi; return u.v;
}

#if defined(__HIP_DEVICE_COMPILE__) && __has_builtin(__builtin_amdgcn_tensor_load_to_lds) && __has_builtin(__builtin_amdgcn_s_wait_tensorcnt)
#define USE_TDM 1
#else
#define USE_TDM 0
#endif

#if USE_TDM
// Issue one TDM load of a [z=3][y=16][x=32] f32 tile of adj_views into LDS.
// remX = columns remaining from tile start (x-OOB reads return 0).
static __device__ __forceinline__ void tdm_load_tile(unsigned lds_off,
                                                     const float* gaddr,
                                                     unsigned remX) {
  unsigned long long ga = (unsigned long long)(uintptr_t)gaddr;
  const unsigned d1s = 36000000u;  // N*N view stride (elements)
  v4u g0;
  g0[0] = 1u;                                    // count=1, user descriptor
  g0[1] = lds_off;                               // lds_addr (bytes)
  g0[2] = (unsigned)(ga & 0xFFFFFFFFu);          // global_addr[31:0]
  g0[3] = (unsigned)((ga >> 32) & 0x01FFFFFFu)   // global_addr[56:32]
          | (2u << 30);                          // type = 2 ("image")
  v8i g1;
  g1[0] = (int)(2u << 16);                       // data_size = 2 -> 4 bytes
  g1[1] = (int)((remX & 0xFFFFu) << 16);         // tensor_dim0[15:0]
  g1[2] = (int)((remX >> 16) | (16u << 16));     // dim0[31:16] | tensor_dim1 lo (16)
  g1[3] = (int)(32u << 16);                      // tensor_dim1 hi | tile_dim0=32
  g1[4] = (int)(16u | (3u << 16));               // tile_dim1=16 | tile_dim2=3
  g1[5] = (int)6000;                             // tensor_dim0_stride lo32
  g1[6] = (int)((d1s & 0xFFFFu) << 16);          // dim0_stride hi16=0 | dim1_stride lo16
  g1[7] = (int)(d1s >> 16);                      // tensor_dim1_stride[47:16]
  v4i g2;
  g2[0] = 3;                                     // tensor_dim2 = 3 views
  g2[1] = 0; g2[2] = 0; g2[3] = 0;
  v4i g3 = {0, 0, 0, 0};
#if defined(__clang_major__) && (__clang_major__ >= 23)
  v8i g4 = {0, 0, 0, 0, 0, 0, 0, 0};
  __builtin_amdgcn_tensor_load_to_lds(g0, g1, g2, g3, g4, 0);
#else
  __builtin_amdgcn_tensor_load_to_lds(g0, g1, g2, g3, 0);
#endif
}
#endif

// ---------------------------------------------------------------------------
// init: g = 1/3 for all (n,v); zero the xT padding columns [6000,6016)
// ---------------------------------------------------------------------------
__global__ void __launch_bounds__(256)
k_init(float* __restrict__ g, __bf16* __restrict__ xT) {
  int i = blockIdx.x * 256 + threadIdx.x;
  if (i < NN6000 * VV) g[i] = (1.0f / 3.0f);
  int j = i - NN6000 * VV;
  if (j >= 0 && j < HH * 16) {
    int h = j >> 4, n = NN6000 + (j & 15);
    xT[(size_t)h * NPAD + n] = (__bf16)0.0f;
  }
}

// ---------------------------------------------------------------------------
// f32 -> bf16 convert with optional column zero-padding
// ---------------------------------------------------------------------------
__global__ void __launch_bounds__(256)
k_cvt(const float* __restrict__ src, __bf16* __restrict__ dst,
      long total, int scols, int dcols) {
  long i = (long)blockIdx.x * 256 + threadIdx.x;
  if (i >= total) return;
  long r = i / dcols; int c = (int)(i % dcols);
  dst[i] = (c < scols) ? (__bf16)src[r * (long)scols + c] : (__bf16)0.0f;
}

// ---------------------------------------------------------------------------
// Generic bf16 WMMA GEMM: out[n,c] = act( sum_k A[n,k]*B[c,k] + bias[c] )
// Block: 256 thr (8 waves). Block tile = 16 rows x 256 cols. Wave: 2 col-tiles.
// mode 0: f32 out (+bias)           (q / k GEMMs)
// mode 1: relu(+bias) -> bf16 out   (per-view projection -> h0s)
// mode 2: x=relu(beta*D+(1-beta)*S) -> bf16 x (row major) + bf16 xT
// ---------------------------------------------------------------------------
__global__ void __launch_bounds__(256)
k_gemm(const __bf16* __restrict__ A, long sA,
       const __bf16* __restrict__ B, long sB,
       const float* __restrict__ bias, long sBias,
       int K, int mode,
       float* __restrict__ outF, long sOut,
       __bf16* __restrict__ outB,
       __bf16* __restrict__ outT,
       const float* __restrict__ Sf, float beta) {
  const int tid = threadIdx.x, lane = tid & 31, w = tid >> 5;
  const int z = blockIdx.z;
  A += (size_t)z * sA;
  B += (size_t)z * sB;
  const float* bptr = bias ? bias + (size_t)z * sBias : (const float*)0;

  const int n0   = blockIdx.x * 16;
  const int rowA = n0 + (lane & 15);
  const int kloA = (lane < 16) ? 0 : 16;   // A: lanes0-15 K0..15
  const int kloB = (lane < 16) ? 0 : 16;   // B: lanes0-15 K0..15
  const int col0 = w * 32 + (lane & 15);

  v8f acc0 = {0,0,0,0,0,0,0,0};
  v8f acc1 = {0,0,0,0,0,0,0,0};

  const __bf16* aRow = A + (size_t)rowA * K;
  const __bf16* bRow0 = B + (size_t)col0 * K;
  const __bf16* bRow1 = B + (size_t)(col0 + 16) * K;

  for (int k = 0; k < K; k += 32) {
    v16bf av = pack16(*(const v8bf*)(aRow + k + kloA),
                      *(const v8bf*)(aRow + k + kloA + 8));
    v16bf bv0 = pack16(*(const v8bf*)(bRow0 + k + kloB),
                       *(const v8bf*)(bRow0 + k + kloB + 8));
    v16bf bv1 = pack16(*(const v8bf*)(bRow1 + k + kloB),
                       *(const v8bf*)(bRow1 + k + kloB + 8));
    acc0 = __builtin_amdgcn_wmma_f32_16x16x32_bf16(false, av, false, bv0,
                                                   (short)0, acc0, false, false);
    acc1 = __builtin_amdgcn_wmma_f32_16x16x32_bf16(false, av, false, bv1,
                                                   (short)0, acc1, false, false);
  }

  const int mBase = (lane >> 4) << 3;  // D: lanes16-31 hold M=8..15
#pragma unroll
  for (int r = 0; r < 8; ++r) {
    int n = n0 + mBase + r;
#pragma unroll
    for (int t = 0; t < 2; ++t) {
      int col = col0 + 16 * t;
      float d = (t == 0) ? acc0[r] : acc1[r];
      if (bptr) d += bptr[col];
      size_t oi = (size_t)n * HH + col;
      if (mode == 0) {
        outF[(size_t)z * sOut + oi] = d;
      } else if (mode == 1) {
        outB[(size_t)z * sOut + oi] = (__bf16)fmaxf(d, 0.0f);
      } else {
        float s = Sf[oi];
        float x = fmaxf(beta * d + (1.0f - beta) * s, 0.0f);
        outB[oi] = (__bf16)x;
        outT[(size_t)col * NPAD + n] = (__bf16)x;
      }
    }
  }
}

// ---------------------------------------------------------------------------
// x0 = relu( softmax(view_w) . h0s ); write bf16 row-major + transposed
// ---------------------------------------------------------------------------
__global__ void __launch_bounds__(256)
k_x0(const __bf16* __restrict__ h0s, const float* __restrict__ vw,
     __bf16* __restrict__ x, __bf16* __restrict__ xT) {
  size_t i = (size_t)blockIdx.x * 256 + threadIdx.x;
  const size_t NH = (size_t)NN6000 * HH;
  if (i >= NH) return;
  float w0 = vw[0], w1 = vw[1], w2 = vw[2];
  float m = fmaxf(w0, fmaxf(w1, w2));
  float e0 = __expf(w0 - m), e1 = __expf(w1 - m), e2 = __expf(w2 - m);
  float inv = 1.0f / (e0 + e1 + e2);
  float v = e0 * inv * (float)h0s[i] + e1 * inv * (float)h0s[NH + i] +
            e2 * inv * (float)h0s[2 * NH + i];
  v = fmaxf(v, 0.0f);
  int n = (int)(i / HH), h = (int)(i % HH);
  x[i] = (__bf16)v;
  xT[(size_t)h * NPAD + n] = (__bf16)v;
}

// ---------------------------------------------------------------------------
// Per-node additive attention over V=3 views (one wave32 per node):
// score_v = sum_h tanh(q+k_v)*av[h] + vb ; att = softmax_v ;
// g = 0.9 g + 0.1 att ; Z0 = sum_v att_v * h0s_v
// ---------------------------------------------------------------------------
__global__ void __launch_bounds__(256)
k_attn(const float* __restrict__ q, const float* __restrict__ kk,
       const __bf16* __restrict__ h0s, const float* __restrict__ av,
       const float* __restrict__ avb, float* __restrict__ g,
       float* __restrict__ Z0) {
  const int lane = threadIdx.x & 31;
  const int n = blockIdx.x * 8 + (threadIdx.x >> 5);
  const size_t NH = (size_t)NN6000 * HH;
  float s0 = 0.f, s1 = 0.f, s2 = 0.f;
#pragma unroll
  for (int j = 0; j < 8; ++j) {
    int h = lane + 32 * j;
    size_t idx = (size_t)n * HH + h;
    float qv = q[idx], a = av[h];
    s0 += tanhf(qv + kk[idx]) * a;
    s1 += tanhf(qv + kk[NH + idx]) * a;
    s2 += tanhf(qv + kk[2 * NH + idx]) * a;
  }
#pragma unroll
  for (int off = 16; off > 0; off >>= 1) {
    s0 += __shfl_xor(s0, off, 32);
    s1 += __shfl_xor(s1, off, 32);
    s2 += __shfl_xor(s2, off, 32);
  }
  float vb = avb[0];
  s0 += vb; s1 += vb; s2 += vb;
  float mx = fmaxf(s0, fmaxf(s1, s2));
  float e0 = __expf(s0 - mx), e1 = __expf(s1 - mx), e2 = __expf(s2 - mx);
  float inv = 1.0f / (e0 + e1 + e2);
  float a0 = e0 * inv, a1 = e1 * inv, a2 = e2 * inv;
  if (lane < 3) {
    float an = (lane == 0) ? a0 : (lane == 1) ? a1 : a2;
    size_t gi = (size_t)n * 3 + lane;
    g[gi] = 0.9f * g[gi] + 0.1f * an;
  }
#pragma unroll
  for (int j = 0; j < 8; ++j) {
    int h = lane + 32 * j;
    size_t idx = (size_t)n * HH + h;
    Z0[idx] = a0 * (float)h0s[idx] + a1 * (float)h0s[NH + idx] +
              a2 * (float)h0s[2 * NH + idx];
  }
}

// ---------------------------------------------------------------------------
// Fused adjacency-blend + SpMM:
// S[n,:] = 0.9 * ( (sum_v g[n,v]*adj0[v,n,:]) @ x ) + 0.1 * Z0[n,:]
// Block = 256 thr; 16-row strip. adj tiles [3][16][32] f32 are DMA'd into
// LDS by the Tensor Data Mover (double buffered, TENSORcnt), blended with g
// into a 16x32 bf16 A tile, then v_wmma_f32_16x16x32_bf16 against xT.
// ---------------------------------------------------------------------------
__global__ void __launch_bounds__(256)
k_spmm(const float* __restrict__ adj, const float* __restrict__ g,
       const __bf16* __restrict__ xT, const float* __restrict__ Z0,
       float* __restrict__ Sf, __bf16* __restrict__ Sb) {
  __shared__ __align__(16) __bf16 At[16 * 32];
  __shared__ float gs[48];
#if USE_TDM
  __shared__ __align__(16) float stage[2][VV * 512];  // [buf][v*512 + r*32 + c]
#endif
  const int tid = threadIdx.x, lane = tid & 31, w = tid >> 5;
  const int n0 = blockIdx.x * 16;
  if (tid < 48) gs[tid] = g[(size_t)(n0 + tid / 3) * 3 + (tid % 3)];

  const size_t NN = (size_t)NN6000 * NN6000;
  const int rowA = lane & 15;
  const int kloA = (lane < 16) ? 0 : 16;
  const int kloB = (lane < 16) ? 0 : 16;
  const int col0 = w * 32 + (lane & 15);
  v8f acc0 = {0,0,0,0,0,0,0,0};
  v8f acc1 = {0,0,0,0,0,0,0,0};

#if USE_TDM
  const float* tileBase = adj + (size_t)n0 * NN6000;
  if (w == 0) {  // wave 0 drives the TDM (EXEC-independent, per-wave issue)
    tdm_load_tile((unsigned)(uintptr_t)&stage[0][0], tileBase, NN6000);
  }
  int p = 0;
  for (int m = 0; m < NN6000; m += 32, p ^= 1) {
    if (w == 0) {
      if (m + 32 < NN6000) {
        tdm_load_tile((unsigned)(uintptr_t)&stage[p ^ 1][0],
                      tileBase + (m + 32), (unsigned)(NN6000 - (m + 32)));
        __builtin_amdgcn_s_wait_tensorcnt(1);  // tile for buf[p] landed
      } else {
        __builtin_amdgcn_s_wait_tensorcnt(0);
      }
    }
    __syncthreads();
    // blend phase (LDS f32 -> bf16 A tile); TDM x-OOB already zero-filled
    const float* sb = stage[p];
    for (int e = tid; e < 512; e += 256) {
      int r = e >> 5;
      float a = gs[r * 3 + 0] * sb[e] +
                gs[r * 3 + 1] * sb[512 + e] +
                gs[r * 3 + 2] * sb[1024 + e];
      At[e] = (__bf16)a;
    }
    __syncthreads();
    v16bf av = pack16(*(const v8bf*)(At + rowA * 32 + kloA),
                      *(const v8bf*)(At + rowA * 32 + kloA + 8));
    const __bf16* bp0 = xT + (size_t)col0 * NPAD + m + kloB;
    const __bf16* bp1 = xT + (size_t)(col0 + 16) * NPAD + m + kloB;
    v16bf bv0 = pack16(*(const v8bf*)bp0, *(const v8bf*)(bp0 + 8));
    v16bf bv1 = pack16(*(const v8bf*)bp1, *(const v8bf*)(bp1 + 8));
    acc0 = __builtin_amdgcn_wmma_f32_16x16x32_bf16(false, av, false, bv0,
                                                   (short)0, acc0, false, false);
    acc1 = __builtin_amdgcn_wmma_f32_16x16x32_bf16(false, av, false, bv1,
                                                   (short)0, acc1, false, false);
    __syncthreads();
  }
#else
  __syncthreads();
  for (int m = 0; m < NN6000; m += 32) {
    for (int e = tid; e < 512; e += 256) {
      int r = e >> 5, c = e & 31, mm = m + c;
      float a = 0.0f;
      if (mm < NN6000) {
        size_t base = (size_t)(n0 + r) * NN6000 + mm;
        a = gs[r * 3 + 0] * adj[base] +
            gs[r * 3 + 1] * adj[NN + base] +
            gs[r * 3 + 2] * adj[2 * NN + base];
        if (c == 0 && mm + 32 < NN6000) {
          __builtin_prefetch(adj + base + 32, 0, 1);
          __builtin_prefetch(adj + NN + base + 32, 0, 1);
          __builtin_prefetch(adj + 2 * NN + base + 32, 0, 1);
        }
      }
      At[e] = (__bf16)a;
    }
    __syncthreads();
    v16bf av = pack16(*(const v8bf*)(At + rowA * 32 + kloA),
                      *(const v8bf*)(At + rowA * 32 + kloA + 8));
    const __bf16* bp0 = xT + (size_t)col0 * NPAD + m + kloB;
    const __bf16* bp1 = xT + (size_t)(col0 + 16) * NPAD + m + kloB;
    v16bf bv0 = pack16(*(const v8bf*)bp0, *(const v8bf*)(bp0 + 8));
    v16bf bv1 = pack16(*(const v8bf*)bp1, *(const v8bf*)(bp1 + 8));
    acc0 = __builtin_amdgcn_wmma_f32_16x16x32_bf16(false, av, false, bv0,
                                                   (short)0, acc0, false, false);
    acc1 = __builtin_amdgcn_wmma_f32_16x16x32_bf16(false, av, false, bv1,
                                                   (short)0, acc1, false, false);
    __syncthreads();
  }
#endif

  const int mBase = (lane >> 4) << 3;
#pragma unroll
  for (int r = 0; r < 8; ++r) {
    int n = n0 + mBase + r;
#pragma unroll
    for (int t = 0; t < 2; ++t) {
      int col = col0 + 16 * t;
      float d = (t == 0) ? acc0[r] : acc1[r];
      size_t oi = (size_t)n * HH + col;
      float s = 0.9f * d + 0.1f * Z0[oi];
      Sf[oi] = s;
      Sb[oi] = (__bf16)s;
    }
  }
}

// ---------------------------------------------------------------------------
// out[n,c] = x[n,:] . Wout[c,:] + bout[c]   (tiny: 6000x7)
// ---------------------------------------------------------------------------
__global__ void __launch_bounds__(256)
k_out(const __bf16* __restrict__ x, const float* __restrict__ Wout,
      const float* __restrict__ bout, float* __restrict__ out) {
  int i = blockIdx.x * 256 + threadIdx.x;
  if (i >= NN6000 * CC) return;
  int n = i / CC, c = i % CC;
  const __bf16* xr = x + (size_t)n * HH;
  const float* wr = Wout + (size_t)c * HH;
  float acc = bout[c];
  for (int h = 0; h < HH; ++h) acc += (float)xr[h] * wr[h];
  out[i] = acc;
}

// ---------------------------------------------------------------------------
extern "C" void kernel_launch(void* const* d_in, const int* in_sizes, int n_in,
                              void* d_out, int out_size, void* d_ws, size_t ws_size,
                              hipStream_t stream) {
  const float* x_views  = (const float*)d_in[0];
  const float* adj      = (const float*)d_in[1];
  const float* W0       = (const float*)d_in[2];
  const float* b0       = (const float*)d_in[3];
  const float* view_w   = (const float*)d_in[4];
  const float* att_Wq   = (const float*)d_in[5];
  const float* att_bq   = (const float*)d_in[6];
  const float* att_Wk   = (const float*)d_in[7];
  const float* att_bk   = (const float*)d_in[8];
  const float* att_v    = (const float*)d_in[9];
  const float* att_vb   = (const float*)d_in[10];
  const float* gc_W     = (const float*)d_in[11];
  const float* Wout     = (const float*)d_in[12];
  const float* bout     = (const float*)d_in[13];
  float* out = (float*)d_out;

  const long N = NN6000, H = HH, V = VV, L = LL;
  const long NH = N * H;

  // workspace carve-up (256B aligned)
  char* base = (char*)d_ws;
  size_t off = 0;
  auto carve = [&](size_t bytes) -> char* {
    char* p = base + off;
    off = (off + bytes + 255) & ~(size_t)255;
    return p;
  };
  __bf16* xv_bf  = (__bf16*)carve(V * N * FP * 2);
  __bf16* W0_bf  = (__bf16*)carve(V * H * FP * 2);
  __bf16* Wq_bf  = (__bf16*)carve(L * H * H * 2);
  __bf16* Wk_bf  = (__bf16*)carve(L * H * H * 2);
  __bf16* gc_bf  = (__bf16*)carve(L * H * H * 2);
  __bf16* h0s_bf = (__bf16*)carve(V * NH * 2);
  __bf16* x_bf   = (__bf16*)carve(NH * 2);
  __bf16* xT_bf  = (__bf16*)carve((size_t)H * NPAD * 2);
  float*  q_ws   = (float*)carve(NH * 4);
  float*  k_ws   = (float*)carve(V * NH * 4);
  float*  g_ws   = (float*)carve(N * 3 * 4);
  float*  Z0_ws  = (float*)carve(NH * 4);
  float*  Sf_ws  = (float*)carve(NH * 4);
  __bf16* Sb_ws  = (__bf16*)carve(NH * 2);

  const int NT = (int)(N / 16);  // 375 row tiles
  dim3 blk(256);

  // init g and xT padding
  {
    int tot = (int)(N * 3 + H * 16);
    k_init<<<dim3((tot + 255) / 256), blk, 0, stream>>>(g_ws, xT_bf);
  }
  // weight / input conversions to bf16 (with F->1024 zero pad)
  {
    long t1 = V * N * FP;
    k_cvt<<<dim3((int)((t1 + 255) / 256)), blk, 0, stream>>>(x_views, xv_bf, t1, 1000, FP);
    long t2 = V * H * FP;
    k_cvt<<<dim3((int)((t2 + 255) / 256)), blk, 0, stream>>>(W0, W0_bf, t2, 1000, FP);
    long t3 = L * H * H;
    k_cvt<<<dim3((int)((t3 + 255) / 256)), blk, 0, stream>>>(att_Wq, Wq_bf, t3, (int)H, (int)H);
    k_cvt<<<dim3((int)((t3 + 255) / 256)), blk, 0, stream>>>(att_Wk, Wk_bf, t3, (int)H, (int)H);
    k_cvt<<<dim3((int)((t3 + 255) / 256)), blk, 0, stream>>>(gc_W, gc_bf, t3, (int)H, (int)H);
  }
  // per-view projection: h0s[v] = relu(xv[v] @ W0[v]^T + b0[v])  (K=1024)
  k_gemm<<<dim3(NT, 1, (int)V), blk, 0, stream>>>(
      xv_bf, N * FP, W0_bf, H * FP, b0, H, FP, 1,
      (float*)0, NH, h0s_bf, (__bf16*)0, (const float*)0, 0.0f);
  // x0 = relu(softmax(view_w) . h0s)
  k_x0<<<dim3((int)((NH + 255) / 256)), blk, 0, stream>>>(h0s_bf, view_w, x_bf, xT_bf);

  for (int i = 0; i < (int)L; ++i) {
    float beta = logf(0.5f / (float)(i + 1) + 1.0f);
    // q = x @ Wq[i]^T + bq[i]
    k_gemm<<<dim3(NT, 1, 1), blk, 0, stream>>>(
        x_bf, 0, Wq_bf + (size_t)i * H * H, 0, att_bq + (size_t)i * H, 0,
        (int)H, 0, q_ws, NH, (__bf16*)0, (__bf16*)0, (const float*)0, 0.0f);
    // k[v] = h0s[v] @ Wk[i]^T + bk[i]
    k_gemm<<<dim3(NT, 1, (int)V), blk, 0, stream>>>(
        h0s_bf, NH, Wk_bf + (size_t)i * H * H, 0, att_bk + (size_t)i * H, 0,
        (int)H, 0, k_ws, NH, (__bf16*)0, (__bf16*)0, (const float*)0, 0.0f);
    // attention, g update, Z0
    k_attn<<<dim3((int)(N / 8)), blk, 0, stream>>>(
        q_ws, k_ws, h0s_bf, att_v + (size_t)i * H, att_vb + i, g_ws, Z0_ws);
    // fused adjacency-blend + SpMM + support blend (TDM-fed)
    k_spmm<<<dim3(NT), blk, 0, stream>>>(adj, g_ws, xT_bf, Z0_ws, Sf_ws, Sb_ws);
    // x = relu(beta * (S @ gcW[i]) + (1-beta) * S)
    k_gemm<<<dim3(NT, 1, 1), blk, 0, stream>>>(
        Sb_ws, 0, gc_bf + (size_t)i * H * H, 0, (const float*)0, 0,
        (int)H, 2, (float*)0, 0, x_bf, xT_bf, Sf_ws, beta);
  }
  // out = x @ Wout^T + bout
  k_out<<<dim3((int)((N * CC + 255) / 256)), blk, 0, stream>>>(x_bf, Wout, bout, out);
  (void)in_sizes; (void)n_in; (void)out_size; (void)ws_size;
}
